// CosformerAttention_5377299054790
// MI455X (gfx1250) — matrix operand order repaired and checked by reading, not compiled, not run on hardware
//
#include <hip/hip_runtime.h>
#include <hip/hip_bf16.h>

typedef __attribute__((ext_vector_type(16))) _Float16 v16h;
typedef __attribute__((ext_vector_type(8)))  float    v8f;
typedef __attribute__((ext_vector_type(4)))  float    v4f;

#define Bq 4
#define Lq 4096
#define Eq 1024
#define Hq 16
#define Dq 64
#define E3 3072
#define EPSq 1e-4f

__device__ __forceinline__ v8f wmma16x16x32(v16h a, v16h b, v8f c) {
  return __builtin_amdgcn_wmma_f32_16x16x32_f16(false, a, false, b, (short)0, c,
                                                false, false);
}

__global__ __launch_bounds__(256) void zero_f32(float* p, int n) {
  int i = blockIdx.x * 256 + threadIdx.x;
  if (i < n) p[i] = 0.f;
}

// C[M,N] = A[M,K] @ B[K,N] + bias[N]  (fp32 global, f16 WMMA compute, f32 acc)
// BM=128 BN=128 BK=32, 256 threads / 8 waves, wave owns 64x32.
// LDS: A is M-major/K-contig; B stored N-major/K-contig so both fragment
// types load as aligned ds_load_b128 (stride 40 halfs = 80B, conflict-free).
// Register double-buffering hides global load latency under WMMA.
__global__ __launch_bounds__(256) void gemm_wmma(const float* __restrict__ A,
                                                 const float* __restrict__ Bm,
                                                 const float* __restrict__ bias,
                                                 float* __restrict__ C,
                                                 int M, int N, int K) {
  __shared__ _Float16 sA[128][40];    // [m][k]
  __shared__ _Float16 sBT[128][40];   // [n][k]
  const int t = threadIdx.x;
  const int lane = t & 31;
  const int wave = t >> 5;
  const int wm = wave >> 2;  // 0..1
  const int wn = wave & 3;   // 0..3
  const int mBase = blockIdx.y * 128;
  const int nBase = blockIdx.x * 128;

  v8f acc[4][2];
#pragma unroll
  for (int mt = 0; mt < 4; ++mt)
#pragma unroll
    for (int nt = 0; nt < 2; ++nt)
#pragma unroll
      for (int i = 0; i < 8; ++i) acc[mt][nt][i] = 0.f;

  const int arow = t >> 1;         // 0..127
  const int acol = (t & 1) * 16;
  const int brow = t >> 3;         // 0..31  (k index)
  const int bcol = (t & 7) * 16;   // n base

  const float* aptr = A + (size_t)(mBase + arow) * K + acol;
  const float* bptr = Bm + (size_t)brow * N + nBase + bcol;

  v4f ra[4], rb[4];
#pragma unroll
  for (int v = 0; v < 4; ++v) {
    ra[v] = *(const v4f*)(aptr + v * 4);
    rb[v] = *(const v4f*)(bptr + v * 4);
  }

  const int nK = K >> 5;
  for (int kt = 0; kt < nK; ++kt) {
    // commit staged registers to LDS (B transposed: n-major, k-contig)
#pragma unroll
    for (int v = 0; v < 4; ++v)
#pragma unroll
      for (int j = 0; j < 4; ++j) {
        sA[arow][acol + v * 4 + j] = (_Float16)ra[v][j];
        sBT[bcol + v * 4 + j][brow] = (_Float16)rb[v][j];
      }
    __syncthreads();

    if (kt + 1 < nK) {  // prefetch next tile into registers (overlaps WMMA)
      const float* an = aptr + (size_t)(kt + 1) * 32;
      const float* bn = bptr + (size_t)(kt + 1) * 32 * N;
#pragma unroll
      for (int v = 0; v < 4; ++v) {
        ra[v] = *(const v4f*)(an + v * 4);
        rb[v] = *(const v4f*)(bn + v * 4);
      }
    }

    v16h af[4];
#pragma unroll
    for (int mt = 0; mt < 4; ++mt) {
      const int m = wm * 64 + mt * 16 + (lane & 15);
      const int kh = (lane >> 4) * 8;
#pragma unroll
      for (int i = 0; i < 8; ++i) {
        af[mt][i] = sA[m][kh + i];
        af[mt][i + 8] = sA[m][16 + kh + i];
      }
    }
#pragma unroll
    for (int nt = 0; nt < 2; ++nt) {
      v16h bf;
      const int n = wn * 32 + nt * 16 + (lane & 15);
      const int kf = (lane >> 4) * 8;
#pragma unroll
      for (int i = 0; i < 16; ++i) bf[i] = sBT[n][kf + (i & 7) + ((i >> 3) << 4)];
#pragma unroll
      for (int mt = 0; mt < 4; ++mt)
        acc[mt][nt] = wmma16x16x32(af[mt], bf, acc[mt][nt]);
    }
    __syncthreads();
  }

#pragma unroll
  for (int nt = 0; nt < 2; ++nt) {
    const int n = nBase + wn * 32 + nt * 16 + (lane & 15);
    const float bv = bias[n];
#pragma unroll
    for (int mt = 0; mt < 4; ++mt)
#pragma unroll
      for (int r = 0; r < 8; ++r) {
        const int m = mBase + wm * 64 + mt * 16 + (lane >> 4) * 8 + r;
        C[(size_t)m * N + n] = acc[mt][nt][r] + bv;
      }
  }
}

// kv_cos[b,h] = k_cos^T @ v, kv_sin likewise; plus per-d column sums.
// grid (seg=4, b*h=64); each block contracts L/4 = 1024 positions, atomics merge.
// K stored d-major (A frags contiguous), V stored e-major (B frags contiguous).
__global__ __launch_bounds__(256) void kv_accum(const float* __restrict__ qkv,
                                                float* __restrict__ kvc,
                                                float* __restrict__ kvs,
                                                float* __restrict__ sumc,
                                                float* __restrict__ sums) {
  __shared__ _Float16 sKc[64][40];  // [d][l]
  __shared__ _Float16 sKs[64][40];  // [d][l]
  __shared__ _Float16 sVT[64][40];  // [e][l]
  __shared__ float sSum[128];

  const int t = threadIdx.x;
  const int lane = t & 31;
  const int wave = t >> 5;
  const int bh = blockIdx.y;
  const int b = bh >> 4, h = bh & 15;
  const int seg = blockIdx.x;

  if (t < 128) sSum[t] = 0.f;

  const int lr = t >> 3;        // 0..31
  const int db = (t & 7) * 8;   // 0..56

  float pc[8], ps[8];
#pragma unroll
  for (int j = 0; j < 8; ++j) { pc[j] = 0.f; ps[j] = 0.f; }

  const int ism = (wave >= 4);
  const int mt = wave & 3;
  v8f acc[4];
#pragma unroll
  for (int nt = 0; nt < 4; ++nt)
#pragma unroll
    for (int i = 0; i < 8; ++i) acc[nt][i] = 0.f;

  const size_t row0 = (size_t)(b * Lq + seg * 1024 + lr) * E3;
  const float* kptr = qkv + row0 + Eq + h * 64 + db;
  const float* vptr = qkv + row0 + 2 * Eq + h * 64 + db;

  v4f rk[2], rv[2];
#pragma unroll
  for (int v = 0; v < 2; ++v) {
    rk[v] = *(const v4f*)(kptr + v * 4);
    rv[v] = *(const v4f*)(vptr + v * 4);
  }

  for (int ch = 0; ch < 32; ++ch) {
    const int l = seg * 1024 + ch * 32 + lr;
    const float ang = (float)l * (1.57079632679f / 4096.0f);
    const float cw = cosf(ang), sw = sinf(ang);
    // commit staged registers: relu + cos/sin weighting, transposed stores
#pragma unroll
    for (int v = 0; v < 2; ++v)
#pragma unroll
      for (int j = 0; j < 4; ++j) {
        const int jj = v * 4 + j;
        const float kr = fmaxf(rk[v][j], 0.f);
        const float kc = kr * cw, ks = kr * sw;
        sKc[db + jj][lr] = (_Float16)kc;
        sKs[db + jj][lr] = (_Float16)ks;
        pc[jj] += kc;
        ps[jj] += ks;
        sVT[db + jj][lr] = (_Float16)rv[v][j];
      }
    __syncthreads();

    if (ch + 1 < 32) {  // prefetch next chunk (32 rows ahead)
      const size_t step = (size_t)(ch + 1) * 32 * E3;
#pragma unroll
      for (int v = 0; v < 2; ++v) {
        rk[v] = *(const v4f*)(kptr + step + v * 4);
        rv[v] = *(const v4f*)(vptr + step + v * 4);
      }
    }

    v16h af;
    {
      const int m = mt * 16 + (lane & 15);
      const int kh = (lane >> 4) * 8;
#pragma unroll
      for (int i = 0; i < 8; ++i) {
        af[i] = ism ? sKs[m][kh + i] : sKc[m][kh + i];
        af[i + 8] = ism ? sKs[m][16 + kh + i] : sKc[m][16 + kh + i];
      }
    }
#pragma unroll
    for (int nt = 0; nt < 4; ++nt) {
      v16h bf;
      const int n = nt * 16 + (lane & 15);
      const int kf = (lane >> 4) * 8;
#pragma unroll
      for (int i = 0; i < 16; ++i) bf[i] = sVT[n][kf + (i & 7) + ((i >> 3) << 4)];
      acc[nt] = wmma16x16x32(af, bf, acc[nt]);
    }
    __syncthreads();
  }

  float* dst = ism ? kvs : kvc;
#pragma unroll
  for (int nt = 0; nt < 4; ++nt) {
    const int e = nt * 16 + (lane & 15);
#pragma unroll
    for (int r = 0; r < 8; ++r) {
      const int d = mt * 16 + (lane >> 4) * 8 + r;
      atomicAdd(&dst[(size_t)bh * 4096 + d * 64 + e], acc[nt][r]);
    }
  }
#pragma unroll
  for (int j = 0; j < 8; ++j) {
    atomicAdd(&sSum[db + j], pc[j]);
    atomicAdd(&sSum[64 + db + j], ps[j]);
  }
  __syncthreads();
  if (t < 64) {
    atomicAdd(&sumc[bh * 64 + t], sSum[t]);
    atomicAdd(&sums[bh * 64 + t], sSum[64 + t]);
  }
}

// context = (q_cos @ kv_cos + q_sin @ kv_sin) / (q_cos.kcs + q_sin.kss + eps)
// grid (L/128, b*h); 8 waves x 16-row strips. KV stored e-major/k-contig in
// LDS (stride 72 halfs = 144B, aligned + conflict-free b128 fragment loads).
__global__ __launch_bounds__(256) void ctx_kernel(const float* __restrict__ qkv,
                                                  const float* __restrict__ kvc,
                                                  const float* __restrict__ kvs,
                                                  const float* __restrict__ sumc,
                                                  const float* __restrict__ sums,
                                                  float* __restrict__ ctx) {
  __shared__ _Float16 sKvC[64][72];  // [n][k]
  __shared__ _Float16 sKvS[64][72];  // [n][k]
  __shared__ float sSC[64], sSS[64];
  __shared__ float sNorm[128];

  const int t = threadIdx.x;
  const int lane = t & 31;
  const int wave = t >> 5;
  const int bh = blockIdx.y;
  const int b = bh >> 4, h = bh & 15;
  const int lt = blockIdx.x;

#pragma unroll
  for (int i = 0; i < 16; ++i) {
    const int f = t * 16 + i;  // 0..4095, row-major [k][n]
    const int k = f >> 6, n = f & 63;
    sKvC[n][k] = (_Float16)kvc[(size_t)bh * 4096 + f];
    sKvS[n][k] = (_Float16)kvs[(size_t)bh * 4096 + f];
  }
  if (t < 64) {
    sSC[t] = sumc[bh * 64 + t];
    sSS[t] = sums[bh * 64 + t];
  }
  __syncthreads();

  if (lane < 16) {  // per-row norm (lanes 0..15 cover the wave's 16 rows)
    const int row = lt * 128 + wave * 16 + lane;
    const float ang = (float)row * (1.57079632679f / 4096.0f);
    const float cw = cosf(ang), sw = sinf(ang);
    const float* qrow = qkv + (size_t)(b * Lq + row) * E3 + h * 64;
    float nrm = 0.f;
#pragma unroll
    for (int d4 = 0; d4 < 16; ++d4) {
      v4f q = *(const v4f*)(qrow + d4 * 4);
#pragma unroll
      for (int j = 0; j < 4; ++j) {
        const float qr = fmaxf(q[j], 0.f);
        nrm += qr * (cw * sSC[d4 * 4 + j] + sw * sSS[d4 * 4 + j]);
      }
    }
    sNorm[wave * 16 + lane] = nrm;
  }
  __syncthreads();

  v16h aC[2], aS[2];
  {
    const int m = lane & 15;
    const int row = lt * 128 + wave * 16 + m;
    const float ang = (float)row * (1.57079632679f / 4096.0f);
    const float cw = cosf(ang), sw = sinf(ang);
    const float* qrow = qkv + (size_t)(b * Lq + row) * E3 + h * 64;
    const int kh = (lane >> 4) * 8;
#pragma unroll
    for (int kc = 0; kc < 2; ++kc)
#pragma unroll
      for (int i = 0; i < 8; ++i) {
        const float q0 = fmaxf(qrow[kc * 32 + kh + i], 0.f);
        const float q1 = fmaxf(qrow[kc * 32 + 16 + kh + i], 0.f);
        aC[kc][i] = (_Float16)(q0 * cw);
        aC[kc][i + 8] = (_Float16)(q1 * cw);
        aS[kc][i] = (_Float16)(q0 * sw);
        aS[kc][i + 8] = (_Float16)(q1 * sw);
      }
  }

#pragma unroll
  for (int nt = 0; nt < 4; ++nt) {
    v8f acc;
#pragma unroll
    for (int i = 0; i < 8; ++i) acc[i] = 0.f;
    const int n = nt * 16 + (lane & 15);
    const int kf = (lane >> 4) * 8;
#pragma unroll
    for (int kc = 0; kc < 2; ++kc) {
      v16h bC, bS;
#pragma unroll
      for (int i = 0; i < 16; ++i) {
        bC[i] = sKvC[n][kc * 32 + kf + (i & 7) + ((i >> 3) << 4)];
        bS[i] = sKvS[n][kc * 32 + kf + (i & 7) + ((i >> 3) << 4)];
      }
      acc = wmma16x16x32(aC[kc], bC, acc);
      acc = wmma16x16x32(aS[kc], bS, acc);
    }
#pragma unroll
    for (int r = 0; r < 8; ++r) {
      const int mrow = (lane >> 4) * 8 + r;
      const int row = lt * 128 + wave * 16 + mrow;
      const float nm = sNorm[wave * 16 + mrow] + EPSq;
      ctx[(size_t)(b * Lq + row) * Eq + h * 64 + n] = acc[r] / nm;
    }
  }
}

extern "C" void kernel_launch(void* const* d_in, const int* in_sizes, int n_in,
                              void* d_out, int out_size, void* d_ws, size_t ws_size,
                              hipStream_t stream) {
  (void)in_sizes; (void)n_in; (void)out_size; (void)ws_size;
  const float* x     = (const float*)d_in[0];
  const float* W_qkv = (const float*)d_in[1];
  const float* b_qkv = (const float*)d_in[2];
  const float* W_out = (const float*)d_in[3];
  const float* b_out = (const float*)d_in[4];
  float* out = (float*)d_out;

  float* qkv = (float*)d_ws;                       // [B,L,3E]   50,331,648 f
  float* ctx = qkv + (size_t)Bq * Lq * E3;         // [B,L,E]    16,777,216 f
  float* kvc = ctx + (size_t)Bq * Lq * Eq;         // [BH,64,64]    262,144 f
  float* kvs = kvc + (size_t)64 * 4096;            // [BH,64,64]    262,144 f
  float* sumc = kvs + (size_t)64 * 4096;           // [BH,64]         4,096 f
  float* sums = sumc + 64 * 64;                    // [BH,64]         4,096 f

  const int nzero = 64 * 4096 * 2 + 64 * 64 * 2;   // kvc..sums contiguous
  zero_f32<<<(nzero + 255) / 256, 256, 0, stream>>>(kvc, nzero);

  gemm_wmma<<<dim3(E3 / 128, (Bq * Lq) / 128), 256, 0, stream>>>(
      x, W_qkv, b_qkv, qkv, Bq * Lq, E3, Eq);
  kv_accum<<<dim3(4, Bq * Hq), 256, 0, stream>>>(qkv, kvc, kvs, sumc, sums);
  ctx_kernel<<<dim3(Lq / 128, Bq * Hq), 256, 0, stream>>>(qkv, kvc, kvs, sumc,
                                                          sums, ctx);
  gemm_wmma<<<dim3(Eq / 128, (Bq * Lq) / 128), 256, 0, stream>>>(
      ctx, W_out, b_out, out, Bq * Lq, Eq, Eq);
}
